// MultiheadAttention_82652350644742
// MI455X (gfx1250) — matrix operand order repaired and checked by reading, not compiled, not run on hardware
//
#include <hip/hip_runtime.h>
#include <hip/hip_bf16.h>

// ---------------------------------------------------------------------------
// MultiheadAttention for MI455X (gfx1250, wave32, WMMA + async-to-LDS DMA).
//   B=2, N=2048, E=1024, H=16, e=64.  All GEMMs on v_wmma_f32_16x16x32_f16.
// ---------------------------------------------------------------------------

typedef _Float16 v16h __attribute__((ext_vector_type(16)));
typedef _Float16 v8h  __attribute__((ext_vector_type(8)));
typedef _Float16 v4h  __attribute__((ext_vector_type(4)));
typedef float    v8f  __attribute__((ext_vector_type(8)));

#define B_   2
#define N_   2048
#define E_   1024
#define H_   16
#define HD_  64            // head dim
#define E3_  3072

// workspace layout (bytes)
#define WS_XH  0u                    // f16 x       [4096][1024]   8 MB
#define WS_WT  (8u  << 20)           // f16 W^T     [3072][1024]   6 MB
#define WS_Q   (14u << 20)           // f16 Q  [B][H][N][64]       8 MB
#define WS_K   (22u << 20)           // f16 K  [B][H][N][64]       8 MB
#define WS_VT  (30u << 20)           // f16 V^T[B][H][64][N]       8 MB

static __device__ __forceinline__ v16h cat16(v8h lo, v8h hi) {
  return __builtin_shufflevector(lo, hi, 0,1,2,3,4,5,6,7,8,9,10,11,12,13,14,15);
}

static __device__ __forceinline__ v8f wmma_f16(v16h a, v16h b, v8f c) {
  return __builtin_amdgcn_wmma_f32_16x16x32_f16(false, a, false, b,
                                                (short)0, c, false, false);
}

// async DMA: global -> LDS, 16 bytes per lane, tracked by ASYNCcnt
static __device__ __forceinline__ void async_b128(unsigned lds_off,
                                                  const void* gptr) {
  unsigned long long g = (unsigned long long)(uintptr_t)gptr;
  asm volatile("global_load_async_to_lds_b128 %0, %1, off"
               :: "v"(lds_off), "v"(g) : "memory");
}
static __device__ __forceinline__ void wait_async0() {
  asm volatile("s_wait_asynccnt 0x0" ::: "memory");
}

// ---------------- pre-pass: fp32 -> f16 (x), fp32 -> f16 transposed (W) -----
__global__ __launch_bounds__(256) void cvt_x_kernel(const float* __restrict__ x,
                                                    _Float16* __restrict__ xh) {
  int i = (blockIdx.x * 256 + threadIdx.x) * 4;
  float4 v = *(const float4*)(x + i);
  v4h o = { (_Float16)v.x, (_Float16)v.y, (_Float16)v.z, (_Float16)v.w };
  *(v4h*)(xh + i) = o;
}

__global__ __launch_bounds__(256) void cvt_wt_kernel(const float* __restrict__ W,
                                                     _Float16* __restrict__ Wt) {
  int i = (blockIdx.x * 256 + threadIdx.x) * 4;   // flat index into W [E][3E]
  int k = i / E3_, c = i % E3_;                   // 3072 % 4 == 0: no row cross
  float4 v = *(const float4*)(W + i);
  Wt[(size_t)(c + 0) * E_ + k] = (_Float16)v.x;
  Wt[(size_t)(c + 1) * E_ + k] = (_Float16)v.y;
  Wt[(size_t)(c + 2) * E_ + k] = (_Float16)v.z;
  Wt[(size_t)(c + 3) * E_ + k] = (_Float16)v.w;
}

// ---------------- kernel 1: QKV projection GEMM + de-interleave -------------
// Block = one (h, s) column group x 8 row-tiles. The 32Kx64-col B slab is
// DMA'd into LDS (double buffered) and shared by all 8 waves.
// Column mapping (reference reshape): flat col c = (e_idx*H + h)*3 + s.
__global__ __launch_bounds__(256) void qkv_gemm_kernel(
    const _Float16* __restrict__ xh, const _Float16* __restrict__ Wt,
    const float* __restrict__ bias,
    _Float16* __restrict__ Qh, _Float16* __restrict__ Kh,
    _Float16* __restrict__ Vt) {
  __shared__ _Float16 Bs[2][64 * 32];              // [buf][col*32 + k]  8 KB

  const int tid = threadIdx.x;
  const int warp = tid >> 5, lane = tid & 31;
  const int hi = lane >> 4, lo = lane & 15;
  const int cg = blockIdx.x % 48;                  // (h, s) column group
  const int rg = blockIdx.x / 48;                  // 0..31
  const int h = cg / 3, s = cg % 3;
  const int rowbase = (rg * 8 + warp) * 16;        // 16-token tile

  // staging: thread -> (col 0..63, 16B chunk 0..3) of the 4 KB B slab
  const int scol = tid >> 2, sc4 = tid & 3;
  const _Float16* sgp = Wt + (size_t)((scol * H_ + h) * 3 + s) * E_ + sc4 * 8;
  const unsigned lbase = (unsigned)(uintptr_t)&Bs[0][0];
  const unsigned sloff = lbase + (unsigned)(scol * 64 + sc4 * 16);

  const _Float16* arow = xh + (size_t)(rowbase + lo) * E_;   // A: lane = row

  async_b128(sloff, sgp);                          // prologue: buf 0, kk = 0

  v8f acc0{}, acc1{}, acc2{}, acc3{};              // fixed accumulator regs
#pragma clang loop unroll(disable)
  for (int it = 0; it < 32; ++it) {
    const int kk = it * 32;
    wait_async0();                                 // buffer (it&1) complete
    __syncthreads();                               // visible to all waves
    if (it < 31)                                   // overlap DMA with WMMAs
      async_b128(sloff + (unsigned)(((it + 1) & 1) * 4096), sgp + kk + 32);

    // A 16x32 f16: elems 0..15 -> K=kk+16*hi.. (contiguous, matches B)
    v16h a = cat16(*(const v8h*)(arow + kk + 16 * hi),
                   *(const v8h*)(arow + kk + 16 * hi + 8));
    const _Float16* bbuf = Bs[it & 1];
    // B 32x16 f16: lane holds col f*16+lo, contiguous K run [16*hi, +16)
    acc0 = wmma_f16(a, *(const v16h*)(bbuf + (0 * 16 + lo) * 32 + 16 * hi), acc0);
    acc1 = wmma_f16(a, *(const v16h*)(bbuf + (1 * 16 + lo) * 32 + 16 * hi), acc1);
    acc2 = wmma_f16(a, *(const v16h*)(bbuf + (2 * 16 + lo) * 32 + 16 * hi), acc2);
    acc3 = wmma_f16(a, *(const v16h*)(bbuf + (3 * 16 + lo) * 32 + 16 * hi), acc3);
  }

  v8f accs[4] = {acc0, acc1, acc2, acc3};
#pragma unroll
  for (int f = 0; f < 4; ++f) {
    const int ecol = f * 16 + lo;
    const int wcol = (ecol * H_ + h) * 3 + s;
    float bv = bias[wcol];
#pragma unroll
    for (int r = 0; r < 8; ++r) {
      int row = rowbase + r + 8 * hi;              // token in [0, B*N)
      int bb = row >> 11, n = row & (N_ - 1);
      _Float16 v = (_Float16)(accs[f][r] + bv);
      size_t head = (size_t)(bb * H_ + h);
      if (s == 0)      Qh[(head * N_ + n) * HD_ + ecol] = v;
      else if (s == 1) Kh[(head * N_ + n) * HD_ + ecol] = v;
      else             Vt[(head * HD_ + ecol) * N_ + n] = v;  // V transposed
    }
  }
}

// ---------------- kernel 2: flash attention -------------------------------
// Block = one (b, h) x 8 row-tiles; the 32-key K and V tiles (4 KB each) are
// DMA'd into LDS (double buffered) and shared by all 8 waves.
__global__ __launch_bounds__(256) void attn_kernel(
    const _Float16* __restrict__ Qh, const _Float16* __restrict__ Kh,
    const _Float16* __restrict__ Vt, float* __restrict__ out) {
  __shared__ _Float16 Ks[2][32 * 64];              // [buf][tokL*64 + e]  8 KB
  __shared__ _Float16 Vs[2][64 * 32];              // [buf][e*32 + tokL]  8 KB
  __shared__ _Float16 pbuf[8][16 * 32];            // wave-private P      8 KB

  const int tid = threadIdx.x;
  const int warp = tid >> 5, lane = tid & 31;
  const int hi = lane >> 4, lo = lane & 15;
  const int bh = blockIdx.x >> 4;                  // (b*H + h) in 0..31
  const int rg = blockIdx.x & 15;
  const int bb = bh >> 4, h = bh & 15;
  const int rowbase = (rg * 8 + warp) * 16;

  const _Float16* kb = Kh + (size_t)bh * N_ * HD_;
  const _Float16* vb = Vt + (size_t)bh * HD_ * N_;

  // Q A-fragments (16 x 64 -> two 16x32 chunks), loaded once from global
  const _Float16* qrow = Qh + ((size_t)bh * N_ + rowbase + lo) * HD_;
  v16h qa0 = cat16(*(const v8h*)(qrow + 16 * hi),
                   *(const v8h*)(qrow + 16 * hi + 8));
  v16h qa1 = cat16(*(const v8h*)(qrow + 32 + 16 * hi),
                   *(const v8h*)(qrow + 32 + 16 * hi + 8));

  // staging: K tile is 4 KB contiguous; V tile is 64 rows x 64 B (stride 4 KB)
  const int svr = tid >> 2, svc = tid & 3;
  const unsigned kL = (unsigned)(uintptr_t)&Ks[0][0] + (unsigned)(tid * 16);
  const unsigned vL = (unsigned)(uintptr_t)&Vs[0][0] +
                      (unsigned)(svr * 64 + svc * 16);
  const _Float16* kg = kb + tid * 8;               // + tok*64
  const _Float16* vg = vb + (size_t)svr * N_ + svc * 8;  // + tok

  async_b128(kL, kg);                              // prologue: buf 0, tok 0
  async_b128(vL, vg);

  v8f o0{}, o1{}, o2{}, o3{};
  float m[8], l[8];
#pragma unroll
  for (int r = 0; r < 8; ++r) { m[r] = -3.0e38f; l[r] = 0.0f; }

#pragma clang loop unroll(disable)
  for (int t = 0; t < 64; ++t) {
    const int tok = t * 32;
    wait_async0();
    __syncthreads();
    if (t < 63) {                                  // overlap next tile's DMA
      unsigned boff = (unsigned)(((t + 1) & 1) * 4096);
      async_b128(kL + boff, kg + (size_t)(tok + 32) * HD_);
      async_b128(vL + boff, vg + tok + 32);
    }
    const _Float16* kt = Ks[t & 1];
    const _Float16* vt = Vs[t & 1];

    // S = Q K^T for two 16-key tiles; B lane = key token, contiguous e-run
    v8f s0{}, s1{};
    s0 = wmma_f16(qa0, *(const v16h*)(kt + (0 * 16 + lo) * 64 + 16 * hi), s0);
    s0 = wmma_f16(qa1, *(const v16h*)(kt + (0 * 16 + lo) * 64 + 32 + 16 * hi), s0);
    s1 = wmma_f16(qa0, *(const v16h*)(kt + (1 * 16 + lo) * 64 + 16 * hi), s1);
    s1 = wmma_f16(qa1, *(const v16h*)(kt + (1 * 16 + lo) * 64 + 32 + 16 * hi), s1);
    s0 = s0 * 0.25f;  // scale = 1/sqrt(num_heads) per reference
    s1 = s1 * 0.25f;

    float alpha[8];
#pragma unroll
    for (int r = 0; r < 8; ++r) {
      float mv = fmaxf(s0[r], s1[r]);              // row r+8*hi
      mv = fmaxf(mv, __shfl_xor(mv, 1));
      mv = fmaxf(mv, __shfl_xor(mv, 2));
      mv = fmaxf(mv, __shfl_xor(mv, 4));
      mv = fmaxf(mv, __shfl_xor(mv, 8));           // reduce over 16 cols
      float mn = fmaxf(m[r], mv);
      alpha[r] = __expf(m[r] - mn);
      m[r] = mn;
      s0[r] = __expf(s0[r] - mn);
      s1[r] = __expf(s1[r] - mn);
      float su = s0[r] + s1[r];
      su += __shfl_xor(su, 1);
      su += __shfl_xor(su, 2);
      su += __shfl_xor(su, 4);
      su += __shfl_xor(su, 8);
      l[r] = l[r] * alpha[r] + su;
    }
#pragma unroll
    for (int r = 0; r < 8; ++r) {
      o0[r] *= alpha[r]; o1[r] *= alpha[r]; o2[r] *= alpha[r]; o3[r] *= alpha[r];
    }

    // P: C/D layout -> A layout via wave-private LDS (16 rows x 32 keys, f16)
    _Float16* pw = pbuf[warp];
#pragma unroll
    for (int r = 0; r < 8; ++r) {
      int row = r + 8 * hi;
      pw[row * 32 + lo]      = (_Float16)s0[r];
      pw[row * 32 + 16 + lo] = (_Float16)s1[r];
    }
    asm volatile("s_wait_dscnt 0" ::: "memory");   // same-wave DS ordering
    const _Float16* pr = pw + lo * 32;
    v16h pa = cat16(*(const v8h*)(pr + 16 * hi),
                    *(const v8h*)(pr + 16 * hi + 8));

    // O += P V : B lane = e-col (V^T row), contiguous token run from LDS
    o0 = wmma_f16(pa, *(const v16h*)(vt + (0 * 16 + lo) * 32 + 16 * hi), o0);
    o1 = wmma_f16(pa, *(const v16h*)(vt + (1 * 16 + lo) * 32 + 16 * hi), o1);
    o2 = wmma_f16(pa, *(const v16h*)(vt + (2 * 16 + lo) * 32 + 16 * hi), o2);
    o3 = wmma_f16(pa, *(const v16h*)(vt + (3 * 16 + lo) * 32 + 16 * hi), o3);
  }

  // normalize and store: out[b, n, h*64 + e]
#pragma unroll
  for (int r = 0; r < 8; ++r) {
    float inv = 1.0f / l[r];
    int n = rowbase + r + 8 * hi;
    size_t base = ((size_t)bb * N_ + n) * E_ + h * HD_ + lo;
    out[base + 0]  = o0[r] * inv;
    out[base + 16] = o1[r] * inv;
    out[base + 32] = o2[r] * inv;
    out[base + 48] = o3[r] * inv;
  }
}

// ---------------------------------------------------------------------------
extern "C" void kernel_launch(void* const* d_in, const int* in_sizes, int n_in,
                              void* d_out, int out_size, void* d_ws, size_t ws_size,
                              hipStream_t stream) {
  const float* x = (const float*)d_in[0];     // [2,2048,1024]
  const float* W = (const float*)d_in[1];     // [1024,3072]
  const float* b = (const float*)d_in[2];     // [3072]
  float* out = (float*)d_out;

  char* ws = (char*)d_ws;
  _Float16* xh = (_Float16*)(ws + WS_XH);
  _Float16* Wt = (_Float16*)(ws + WS_WT);
  _Float16* Qh = (_Float16*)(ws + WS_Q);
  _Float16* Kh = (_Float16*)(ws + WS_K);
  _Float16* Vt = (_Float16*)(ws + WS_VT);

  cvt_x_kernel<<<(B_ * N_ * E_) / (256 * 4), 256, 0, stream>>>(x, xh);
  cvt_wt_kernel<<<(E_ * E3_) / (256 * 4), 256, 0, stream>>>(W, Wt);
  // 48 (h,s) column groups x 32 row groups
  qkv_gemm_kernel<<<48 * 32, 256, 0, stream>>>(xh, Wt, b, Qh, Kh, Vt);
  // 32 (b,h) x 16 row groups
  attn_kernel<<<32 * 16, 256, 0, stream>>>(Qh, Kh, Vt, out);
}
